// SimpleSimilarityMatcher_57526791962712
// MI455X (gfx1250) — matrix operand order repaired and checked by reading, not compiled, not run on hardware
//
#include <hip/hip_runtime.h>

typedef __attribute__((ext_vector_type(16))) _Float16 v16h;
typedef __attribute__((ext_vector_type(8)))  _Float16 v8h;
typedef __attribute__((ext_vector_type(8)))  float    v8f;

#define B_   16
#define S1_  2048
#define S2_  2048
#define D_   1024
#define H_   512
#define SCALE_ 0.044194173824159216f  /* 1/sqrt(512) */

__device__ __forceinline__ v8f wmma_f16(v16h a, v16h b, v8f c) {
  // D = A(16x32 f16) x B(32x16 f16) + C(16x16 f32)
  return __builtin_amdgcn_wmma_f32_16x16x32_f16(false, a, false, b, (short)0, c,
                                                false, false);
}

__device__ __forceinline__ v8f zero8() {
  v8f z = {0.f,0.f,0.f,0.f,0.f,0.f,0.f,0.f};
  return z;
}

__device__ __forceinline__ v16h load_afrag(const _Float16* qrow, int kk, int hi) {
  // A 16x32 f16 fragment row slice: two 16B chunks per lane
  v8h lo  = *(const v8h*)(qrow + kk*32 + hi*16);
  v8h hiv = *(const v8h*)(qrow + kk*32 + hi*16 + 8);
  v16h a;
  #pragma unroll
  for (int j = 0; j < 8; ++j) { a[j] = lo[j]; a[8+j] = hiv[j]; }
  return a;
}

// ---------------------------------------------------------------------------
// K-1: one-time f32 -> f16 conversion of a weight matrix (H*D elements)
// ---------------------------------------------------------------------------
__global__ void __launch_bounds__(256)
cvt_f16_kernel(const float* __restrict__ src, _Float16* __restrict__ dst, int n) {
  const int i = blockIdx.x * 256 + threadIdx.x;
  if (i < n) dst[i] = (_Float16)src[i];
}

// ---------------------------------------------------------------------------
// K0: Y[b,s,h] = (f16) sum_d X[b,s,d]*Wh[h,d] + bias[h]   (Wh already f16)
// block = 256 (8 waves). Wave tile = 16 rows x 128 cols (8 accumulators).
// All 8 B fragments loaded up front per K-step so loads overlap the WMMAs.
// ---------------------------------------------------------------------------
__global__ void __launch_bounds__(256)
proj_f16_kernel(const float* __restrict__ X, const _Float16* __restrict__ Wh,
                const float* __restrict__ bias, _Float16* __restrict__ Y, int S) {
  const int lane = threadIdx.x & 31;
  const int wave = threadIdx.x >> 5;
  const int n  = lane & 15;    // A: row M; B: col N; C: col N
  const int hi = lane >> 4;
  const int b  = blockIdx.y;
  const int s0 = blockIdx.x * 32 + (wave >> 2) * 16;
  const int h0 = (wave & 3) * 128;

  const float* Xrow = X + (size_t)b * S * D_ + (size_t)(s0 + n) * D_;
  const _Float16* Wrow = Wh + (size_t)n * D_ + hi*16;  // + (h0+t*16)*D_ + d0

  v8f acc[8];
  #pragma unroll
  for (int t = 0; t < 8; ++t) acc[t] = zero8();

  for (int d0 = 0; d0 < D_; d0 += 32) {
    // All 8 B fragments first: distinct registers -> one load clause,
    // waits retire incrementally while the WMMA chain executes.
    v16h bf[8];
    #pragma unroll
    for (int t = 0; t < 8; ++t)
      bf[t] = *(const v16h*)(Wrow + (size_t)(h0 + t*16) * D_ + d0);

    v16h a;
    #pragma unroll
    for (int j = 0; j < 8; ++j) a[j]     = (_Float16)Xrow[d0 + hi*16 + j];
    #pragma unroll
    for (int j = 0; j < 8; ++j) a[8 + j] = (_Float16)Xrow[d0 + hi*16 + 8 + j];

    #pragma unroll
    for (int t = 0; t < 8; ++t) acc[t] = wmma_f16(a, bf[t], acc[t]);
  }

  _Float16* Yb = Y + (size_t)b * S * H_;
  #pragma unroll
  for (int t = 0; t < 8; ++t) {
    const int h  = h0 + t*16 + n;
    const float bs = bias[h];
    #pragma unroll
    for (int r = 0; r < 8; ++r) {
      const int row = s0 + r + 8*hi;
      Yb[(size_t)row * H_ + h] = (_Float16)(acc[t][r] + bs);
    }
  }
}

// ---------------------------------------------------------------------------
// K1: online softmax stats. Wave owns 16 q-rows; loops over all t tiles.
// m[b,s] = max_t score, L[b,s] = sum_t exp(score - m)
// K-loop double-buffers k B-fragments in groups of 4.
// ---------------------------------------------------------------------------
__global__ void __launch_bounds__(256)
attn_stats_kernel(const _Float16* __restrict__ Q, const _Float16* __restrict__ Kf,
                  float* __restrict__ Mout, float* __restrict__ Lout) {
  const int lane = threadIdx.x & 31;
  const int wave = threadIdx.x >> 5;
  const int n  = lane & 15;
  const int hi = lane >> 4;
  const int b  = blockIdx.y;
  const int s0 = blockIdx.x * 128 + wave * 16;

  const _Float16* Qb = Q  + (size_t)b * S1_ * H_;
  const _Float16* Kb = Kf + (size_t)b * S2_ * H_;

  // Hoist the 16 q A-fragments (rows s0..s0+15, K = 0..511)
  v16h qf[16];
  {
    const _Float16* qrow = Qb + (size_t)(s0 + n) * H_;
    #pragma unroll
    for (int kk = 0; kk < 16; ++kk) qf[kk] = load_afrag(qrow, kk, hi);
  }

  float mrun[8], lrun[8];
  #pragma unroll
  for (int r = 0; r < 8; ++r) { mrun[r] = -3.0e38f; lrun[r] = 0.f; }

  for (int t0 = 0; t0 < S2_; t0 += 16) {
    const _Float16* krow = Kb + (size_t)(t0 + n) * H_ + hi*16;
    v8f c = zero8();

    v16h kb0[4], kb1[4];
    #pragma unroll
    for (int j = 0; j < 4; ++j) kb0[j] = *(const v16h*)(krow + j*32);
    #pragma unroll
    for (int g = 0; g < 4; ++g) {
      if (g & 1) {
        if (g < 3) {
          #pragma unroll
          for (int j = 0; j < 4; ++j) kb0[j] = *(const v16h*)(krow + ((g+1)*4 + j)*32);
        }
        #pragma unroll
        for (int j = 0; j < 4; ++j) c = wmma_f16(qf[g*4 + j], kb1[j], c);
      } else {
        if (g < 3) {
          #pragma unroll
          for (int j = 0; j < 4; ++j) kb1[j] = *(const v16h*)(krow + ((g+1)*4 + j)*32);
        }
        #pragma unroll
        for (int j = 0; j < 4; ++j) c = wmma_f16(qf[g*4 + j], kb0[j], c);
      }
    }

    #pragma unroll
    for (int r = 0; r < 8; ++r) {
      float x = c[r] * SCALE_;
      float tm = x;
      tm = fmaxf(tm, __shfl_xor(tm, 1, 32));
      tm = fmaxf(tm, __shfl_xor(tm, 2, 32));
      tm = fmaxf(tm, __shfl_xor(tm, 4, 32));
      tm = fmaxf(tm, __shfl_xor(tm, 8, 32));
      const float nm = fmaxf(mrun[r], tm);
      float e = __expf(x - nm);
      e += __shfl_xor(e, 1, 32);
      e += __shfl_xor(e, 2, 32);
      e += __shfl_xor(e, 4, 32);
      e += __shfl_xor(e, 8, 32);
      lrun[r] = lrun[r] * __expf(mrun[r] - nm) + e;
      mrun[r] = nm;
    }
  }

  if (n == 0) {  // lane 0 writes rows 0..7, lane 16 writes rows 8..15
    #pragma unroll
    for (int r = 0; r < 8; ++r) {
      const int row = s0 + r + 8*hi;
      Mout[(size_t)b * S1_ + row] = mrun[r];
      Lout[(size_t)b * S1_ + row] = lrun[r];
    }
  }
}

// ---------------------------------------------------------------------------
// K2: w[b,t] = sum_s exp(score[s,t]-m_s)/L_s. Wave owns 16 t-cols; loops s.
// K-loop double-buffers q A-fragments in groups of 4.
// ---------------------------------------------------------------------------
__global__ void __launch_bounds__(256)
attn_weights_kernel(const _Float16* __restrict__ Q, const _Float16* __restrict__ Kf,
                    const float* __restrict__ Marr, const float* __restrict__ Larr,
                    float* __restrict__ Wout) {
  const int lane = threadIdx.x & 31;
  const int wave = threadIdx.x >> 5;
  const int n  = lane & 15;
  const int hi = lane >> 4;
  const int b  = blockIdx.y;
  const int t0 = blockIdx.x * 128 + wave * 16;

  const _Float16* Qb = Q  + (size_t)b * S1_ * H_;
  const _Float16* Kb = Kf + (size_t)b * S2_ * H_;

  // Hoist the 16 k B-fragments for this t-tile (N = t0+n, contiguous K loads)
  v16h kfr[16];
  {
    const _Float16* krow = Kb + (size_t)(t0 + n) * H_ + hi*16;
    #pragma unroll
    for (int kk = 0; kk < 16; ++kk) kfr[kk] = *(const v16h*)(krow + kk*32);
  }

  float wacc = 0.f;
  for (int s0 = 0; s0 < S1_; s0 += 16) {
    const _Float16* qrow = Qb + (size_t)(s0 + n) * H_;
    v8f c = zero8();

    v16h qa0[4], qa1[4];
    #pragma unroll
    for (int j = 0; j < 4; ++j) qa0[j] = load_afrag(qrow, j, hi);
    #pragma unroll
    for (int g = 0; g < 4; ++g) {
      if (g & 1) {
        if (g < 3) {
          #pragma unroll
          for (int j = 0; j < 4; ++j) qa0[j] = load_afrag(qrow, (g+1)*4 + j, hi);
        }
        #pragma unroll
        for (int j = 0; j < 4; ++j) c = wmma_f16(qa1[j], kfr[g*4 + j], c);
      } else {
        if (g < 3) {
          #pragma unroll
          for (int j = 0; j < 4; ++j) qa1[j] = load_afrag(qrow, (g+1)*4 + j, hi);
        }
        #pragma unroll
        for (int j = 0; j < 4; ++j) c = wmma_f16(qa0[j], kfr[g*4 + j], c);
      }
    }

    float part = 0.f;
    #pragma unroll
    for (int r = 0; r < 8; ++r) {
      const int row = s0 + r + 8*hi;
      const float ms = Marr[(size_t)b * S1_ + row];
      const float lv = Larr[(size_t)b * S1_ + row];
      part += __expf(c[r] * SCALE_ - ms) / lv;
    }
    wacc += part;
  }
  wacc += __shfl_xor(wacc, 16, 32);  // combine rows 0-7 and 8-15 halves
  if (lane < 16) Wout[(size_t)b * S2_ + t0 + lane] = wacc;
}

// ---------------------------------------------------------------------------
// K3: out[b,d] = sum_t w[b,t] * X[b,t,d]   (w == nullptr -> plain column sum)
// ---------------------------------------------------------------------------
__global__ void __launch_bounds__(256)
wcolsum_kernel(const float* __restrict__ X, const float* __restrict__ w,
               float* __restrict__ out, int S) {
  const int b = blockIdx.y;
  const int d = blockIdx.x * 256 + threadIdx.x;
  const float* Xb = X + (size_t)b * S * D_;
  float acc = 0.f;
  if (w) {
    const float* wb = w + (size_t)b * S;
    for (int t = 0; t < S; ++t) acc += wb[t] * Xb[(size_t)t * D_ + d];
  } else {
    for (int t = 0; t < S; ++t) acc += Xb[(size_t)t * D_ + d];
  }
  out[(size_t)b * D_ + d] = acc;
}

// ---------------------------------------------------------------------------
// K4: pooled[b,0:H]  = (sum_d cs1[b,d]*Wq[h,d])/S1 + bq[h]
//     pooled[b,H:2H] = (sum_d u[b,d]  *Wv[h,d])/S1 + bv[h]
// ---------------------------------------------------------------------------
__global__ void __launch_bounds__(256)
pool_kernel(const float* __restrict__ cs1, const float* __restrict__ u,
            const float* __restrict__ Wq, const float* __restrict__ bq,
            const float* __restrict__ Wv, const float* __restrict__ bv,
            float* __restrict__ pooled) {
  const int idx  = blockIdx.x * 256 + threadIdx.x;  // 2*B*H total
  const int half = idx / (B_ * H_);
  const int rem  = idx % (B_ * H_);
  const int b = rem / H_, h = rem % H_;
  const float inv = 1.0f / (float)S1_;
  const float* vec = (half == 0) ? (cs1 + (size_t)b * D_) : (u + (size_t)b * D_);
  const float* wr  = (half == 0) ? (Wq + (size_t)h * D_)  : (Wv + (size_t)h * D_);
  float acc = 0.f;
  for (int d = 0; d < D_; ++d) acc += vec[d] * wr[d];
  const float bias = (half == 0) ? bq[h] : bv[h];
  pooled[(size_t)b * D_ + half * H_ + h] = acc * inv + bias;
}

// ---------------------------------------------------------------------------
// K5: classifier: out[b] = relu(pooled @ Wc1^T + bc1) @ Wc2^T + bc2
// ---------------------------------------------------------------------------
__global__ void __launch_bounds__(512)
cls_kernel(const float* __restrict__ pooled, const float* __restrict__ Wc1,
           const float* __restrict__ bc1, const float* __restrict__ Wc2,
           const float* __restrict__ bc2, float* __restrict__ out) {
  __shared__ float sm[512];
  const int b = blockIdx.x, j = threadIdx.x;
  const float* p  = pooled + (size_t)b * D_;
  const float* wr = Wc1 + (size_t)j * D_;
  float acc = 0.f;
  for (int d = 0; d < D_; ++d) acc += p[d] * wr[d];
  acc = fmaxf(acc + bc1[j], 0.f);
  sm[j] = acc * Wc2[j];
  __syncthreads();
  for (int st = 256; st > 0; st >>= 1) {
    if (j < st) sm[j] += sm[j + st];
    __syncthreads();
  }
  if (j == 0) out[b] = sm[0] + bc2[0];
}

// ---------------------------------------------------------------------------
extern "C" void kernel_launch(void* const* d_in, const int* in_sizes, int n_in,
                              void* d_out, int out_size, void* d_ws, size_t ws_size,
                              hipStream_t stream) {
  const float* f1  = (const float*)d_in[0];
  const float* f2  = (const float*)d_in[1];
  const float* Wq  = (const float*)d_in[2];
  const float* bq  = (const float*)d_in[3];
  const float* Wk  = (const float*)d_in[4];
  const float* bk  = (const float*)d_in[5];
  const float* Wv  = (const float*)d_in[6];
  const float* bv  = (const float*)d_in[7];
  const float* Wc1 = (const float*)d_in[8];
  const float* bc1 = (const float*)d_in[9];
  const float* Wc2 = (const float*)d_in[10];
  const float* bc2 = (const float*)d_in[11];
  float* out = (float*)d_out;

  char* ws = (char*)d_ws;
  size_t off = 0;
  auto carve = [&](size_t bytes) -> void* {
    void* p = (void*)(ws + off);
    off += (bytes + 255) & ~(size_t)255;
    return p;
  };
  _Float16* qh   = (_Float16*)carve((size_t)B_ * S1_ * H_ * 2);   // 33.5 MB
  _Float16* kh   = (_Float16*)carve((size_t)B_ * S2_ * H_ * 2);   // 33.5 MB
  _Float16* wqh  = (_Float16*)carve((size_t)H_ * D_ * 2);         // 1 MB
  _Float16* wkh  = (_Float16*)carve((size_t)H_ * D_ * 2);         // 1 MB
  float*    marr = (float*)carve((size_t)B_ * S1_ * 4);
  float*    larr = (float*)carve((size_t)B_ * S1_ * 4);
  float*    warr = (float*)carve((size_t)B_ * S2_ * 4);
  float*    cs1  = (float*)carve((size_t)B_ * D_ * 4);
  float*    u    = (float*)carve((size_t)B_ * D_ * 4);
  float*    pooled = (float*)carve((size_t)B_ * D_ * 4);
  (void)ws_size; (void)in_sizes; (void)n_in; (void)out_size;

  const int nw = H_ * D_;  // 524288 weight elements

  // One-time weight conversion f32 -> f16
  cvt_f16_kernel<<<dim3(nw / 256), 256, 0, stream>>>(Wq, wqh, nw);
  cvt_f16_kernel<<<dim3(nw / 256), 256, 0, stream>>>(Wk, wkh, nw);

  // Projections q, k (f16 for WMMA); 32-row blocks, 16x128 wave tiles
  proj_f16_kernel<<<dim3(S1_/32, B_), 256, 0, stream>>>(f1, wqh, bq, qh, S1_);
  proj_f16_kernel<<<dim3(S2_/32, B_), 256, 0, stream>>>(f2, wkh, bk, kh, S2_);

  // Column sum of features1 (for mean of q without re-reading q)
  wcolsum_kernel<<<dim3(D_/256, B_), 256, 0, stream>>>(f1, nullptr, cs1, S1_);

  // Softmax stats (m, L) per q-row
  attn_stats_kernel<<<dim3(S1_/128, B_), 256, 0, stream>>>(qh, kh, marr, larr);

  // Attention column weights w[b,t]
  attn_weights_kernel<<<dim3(S2_/128, B_), 256, 0, stream>>>(qh, kh, marr, larr, warr);

  // u[b,d] = sum_t w[b,t] * features2[b,t,d]
  wcolsum_kernel<<<dim3(D_/256, B_), 256, 0, stream>>>(f2, warr, u, S2_);

  // pooled (B, D): [mean q | mean attended]
  pool_kernel<<<dim3((2*B_*H_)/256), 256, 0, stream>>>(cs1, u, Wq, bq, Wv, bv, pooled);

  // classifier MLP -> (B, 1)
  cls_kernel<<<dim3(B_), 512, 0, stream>>>(pooled, Wc1, bc1, Wc2, bc2, out);
}
